// GraphAttentionNetwork_44332652429421
// MI455X (gfx1250) — matrix operand order repaired and checked by reading, not compiled, not run on hardware
//
#include <hip/hip_runtime.h>
#include <math.h>

typedef __attribute__((ext_vector_type(16))) _Float16 v16h;
typedef __attribute__((ext_vector_type(8)))  float    v8f;

#define NNODES 4096
#define NEDGES 131072
#define DMODEL 256
#define NHEADS 8
#define DHEAD  32
#define NEGV   (-1.0e30f)

// ---------------------------------------------------------------------------
// CDNA5 16x16x32 f16 WMMA wave32 layouts (ISA 7.12.2):
// A 16x32 (MxK): row = lane&15; VGPR v holds K pair k0,k0+1,
//   k0 = (v<4?0:16) + (lane>=16?8:0) + (v&3)*2
// B 32x16 (KxN): n = lane&15; hw k order as A; this file packs A and B alike: VGPR v holds K pair kb,kb+1,
//   kb = (lane>=16?16:0) + 2v          (within-half offset = element index 2v+j)
// C/D 16x16 f32: M = r + 8*(lane>=16), N = lane&15
//
// "Packed fragment" format: [tile][lane][16 halfs] = 32 contiguous bytes per
// lane -> one v16h (2x global_load_b128 / ds_load_b128) per WMMA operand.
// ---------------------------------------------------------------------------

__device__ __forceinline__ v16h load_a_frag_f32(const float* __restrict__ A,
                                                int lda, int m0, int kk, int lane) {
  int row = lane & 15, hf = lane >> 4;
  v16h a;
#pragma unroll
  for (int v = 0; v < 8; ++v) {
    int k0 = (hf ? 16 : 0) + v * 2;
    float2 t = *(const float2*)(A + (size_t)(m0 + row) * lda + kk + k0);
    a[2 * v]     = (_Float16)t.x;
    a[2 * v + 1] = (_Float16)t.y;
  }
  return a;
}

// ---------------------------------------------------------------------------
// Pack fp32 weight B[K,N] into GEMM B-fragment format, one wave per 32x16 tile
// ---------------------------------------------------------------------------
__global__ __launch_bounds__(128)
void pack_b_kernel(const float* __restrict__ B, _Float16* __restrict__ Bp,
                   int K, int N) {
  int lane = threadIdx.x & 31;
  int wave = threadIdx.x >> 5;
  int tile = blockIdx.x * 4 + wave;
  int ntN = N >> 4;
  int kt = tile / ntN, nt = tile % ntN;
  int n = lane & 15, hf = lane >> 4;
  v16h b;
#pragma unroll
  for (int v = 0; v < 8; ++v) {
    int kb = (hf ? 16 : 0) + 2 * v;
    b[2 * v]     = (_Float16)B[(size_t)(kt * 32 + kb) * N + nt * 16 + n];
    b[2 * v + 1] = (_Float16)B[(size_t)(kt * 32 + kb + 1) * N + nt * 16 + n];
  }
  *(v16h*)(Bp + (((size_t)tile * 32 + lane) << 4)) = b;
}

// ---------------------------------------------------------------------------
// GEMM: act(A[M,K] @ B + bias), A fp32, B pre-packed f16 fragments.
// One wave computes a 16x64 tile (4 accumulators, A fragment reused 4x).
// Epilogue modes:
//  0: fp32 out                      1: fp32 out + ELU
//  2: Q pack  (A-frag [h][rowtile][lane][16], value*scale)
//  3: K pack  (B-frag over dh  [h][keytile][lane][16])
//  4: V pack  (B-frag over key [h][dhhalf][keytile32][lane][16])
//  5: weight B pack (GEMM B-frag [kt][nt][lane][16])
// ---------------------------------------------------------------------------
__global__ __launch_bounds__(128)
void gemm_wmma_kernel(const float* __restrict__ A, const _Float16* __restrict__ Bp,
                      const float* __restrict__ bias, float* __restrict__ Cf,
                      _Float16* __restrict__ Ch,
                      int M, int K, int N, int mode, float scale) {
  int lane = threadIdx.x & 31;
  int wave = threadIdx.x >> 5;
  int tile = blockIdx.x * 4 + wave;
  int tilesM = M >> 4;
  int tm = tile % tilesM;
  int tn = tile / tilesM;
  int m0 = tm * 16, n0 = tn * 64;
  int ntN = N >> 4;

  v8f acc[4] = {{}, {}, {}, {}};
  for (int kk = 0; kk < K; kk += 32) {
    v16h a = load_a_frag_f32(A, K, m0, kk, lane);
    int kt = kk >> 5;
#pragma unroll
    for (int s = 0; s < 4; ++s) {
      int nt = (n0 >> 4) + s;
      v16h b = *(const v16h*)(Bp + ((((size_t)kt * ntN + nt) * 32 + lane) << 4));
      acc[s] = __builtin_amdgcn_wmma_f32_16x16x32_f16(false, a, false, b,
                                                      (short)0, acc[s], false, false);
    }
  }

  int n = lane & 15, hf = lane >> 4;
#pragma unroll
  for (int s = 0; s < 4; ++s) {
    int col = n0 + s * 16 + n;
    float bv = bias ? bias[col] : 0.f;
#pragma unroll
    for (int r = 0; r < 8; ++r) {
      int row = m0 + r + 8 * hf;
      float val = acc[s][r] + bv;
      if (mode == 0) {
        Cf[(size_t)row * N + col] = val;
      } else if (mode == 1) {
        Cf[(size_t)row * N + col] = (val > 0.f) ? val : expm1f(val);
      } else if (mode == 2) {         // Q -> A-fragment pack (scaled)
        int hh = col >> 5, d = col & 31;
        int p = d >> 1, j = d & 1;
        int lane2 = (row & 15) + (((p >> 3) & 1) << 4);
        int v2 = p & 7;
        Ch[((((size_t)hh * (M >> 4) + (row >> 4)) * 32 + lane2) << 4) + 2 * v2 + j]
            = (_Float16)(val * scale);
      } else if (mode == 3) {         // K -> B-fragment pack over dh
        int hh = col >> 5, d = col & 31;
        int lane2 = ((d >> 4) << 4) + (row & 15);
        Ch[((((size_t)hh * (M >> 4) + (row >> 4)) * 32 + lane2) << 4) + (d & 15)]
            = (_Float16)val;
      } else if (mode == 4) {         // V -> B-fragment pack over keys
        int hh = col >> 5, d = col & 31;
        int td = d >> 4, n2 = d & 15;
        int kloc = row & 31;
        int lane2 = ((kloc >> 4) << 4) + n2;
        Ch[(((((size_t)hh * 2 + td) * (M >> 5) + (row >> 5)) * 32 + lane2) << 4)
           + (kloc & 15)] = (_Float16)val;
      } else {                        // mode 5: weight -> GEMM B-fragment pack
        int nt = col >> 4, n2 = col & 15;
        int kloc = row & 31;
        int lane2 = ((kloc >> 4) << 4) + n2;
        Ch[((((size_t)(row >> 5) * ntN + nt) * 32 + lane2) << 4) + (kloc & 15)]
            = (_Float16)val;
      }
    }
  }
}

// ---------------------------------------------------------------------------
// Prep kernels
// ---------------------------------------------------------------------------
__global__ void softplus_kernel(const float* __restrict__ w, float* __restrict__ ew) {
  int t = threadIdx.x;
  if (t < 8) {
    float x = w[t];
    ew[t] = (x > 20.f) ? x : log1pf(__expf(x));
  }
}

__global__ void clear_mask_kernel(unsigned int* __restrict__ m) {
  size_t i = (size_t)blockIdx.x * blockDim.x + threadIdx.x;
  m[i] = 0u;
}

__global__ void scatter_edges_kernel(const int* __restrict__ ei,
                                     const int* __restrict__ et,
                                     unsigned char* __restrict__ mask) {
  int e = blockIdx.x * blockDim.x + threadIdx.x;
  int s = ei[e];
  int d = ei[NEDGES + e];
  mask[(size_t)s * NNODES + d] = (unsigned char)et[e];   // 1..4
}

__global__ void build_xc_kernel(const float* __restrict__ agg,
                                const int* __restrict__ sid,
                                const float* __restrict__ spk,
                                float* __restrict__ xc) {
  int i = blockIdx.x;
  int t = threadIdx.x;
  xc[(size_t)i * 512 + t] = agg[(size_t)i * 256 + t];
  int s = sid[i];
  xc[(size_t)i * 512 + 256 + t] = spk[(size_t)s * 256 + t];
}

// bf[j] = b2[j] + sum_i b1[i] * W2[i][j]
__global__ void fused_bias_kernel(const float* __restrict__ b1,
                                  const float* __restrict__ W2,
                                  const float* __restrict__ b2,
                                  float* __restrict__ out) {
  int j = threadIdx.x;   // 256 threads
  float s = b2[j];
  for (int i = 0; i < 256; ++i) s += b1[i] * W2[(size_t)i * 256 + j];
  out[j] = s;
}

// ---------------------------------------------------------------------------
// Flash attention: one wave per (head, 32-query block). All WMMA operands are
// single v16h loads from packed fragments (Q/K/V global, P via LDS A-fragment).
// Online softmax: one query row per lane. Mask: byte array (0 = no edge ->
// NEG / diagonal rule; 1..4 -> softplus(edge_w[type-1])).
// ---------------------------------------------------------------------------
__global__ __launch_bounds__(128)
void attn_kernel(const _Float16* __restrict__ qp, const _Float16* __restrict__ kp,
                 const _Float16* __restrict__ vp, const unsigned char* __restrict__ mask,
                 const float* __restrict__ ew, float* __restrict__ out) {
  __shared__ float s_tile[4][32][32];
  __shared__ v16h  p_frag[4][2][32];
  __shared__ float stat[4][32];

  int lane = threadIdx.x & 31;
  int w = threadIdx.x >> 5;
  int gwave = blockIdx.x * 4 + w;
  int h = gwave >> 7;            // 128 query blocks per head
  int qb = gwave & 127;
  int qbase = qb * 32;
  int n = lane & 15, hf = lane >> 4;

  float ew0 = ew[0], ew1 = ew[1], ew2 = ew[2], ew3 = ew[3];

  const v16h* qv = (const v16h*)qp;
  const v16h* kv = (const v16h*)kp;
  const v16h* vv = (const v16h*)vp;

  v16h qa0 = qv[((size_t)h * (NNODES / 16) + qb * 2)     * 32 + lane];
  v16h qa1 = qv[((size_t)h * (NNODES / 16) + qb * 2 + 1) * 32 + lane];

  v8f o00 = {}, o01 = {}, o10 = {}, o11 = {};  // o[qtile][dh-half]
  float m_run = -__builtin_inff();
  float l_run = 0.f;

  for (int kb = 0; kb < NNODES; kb += 32) {
    if (kb + 32 < NNODES) {   // prefetch next K/V fragments
      __builtin_prefetch(&kv[((size_t)h * (NNODES / 16) + ((kb + 32) >> 4)) * 32 + lane], 0, 0);
      __builtin_prefetch(&vv[((size_t)h * 2 * (NNODES / 32) + ((kb + 32) >> 5)) * 32 + lane], 0, 0);
    }
    // ---- scores: S[32x32] = Q @ K^T ----
#pragma unroll
    for (int tk = 0; tk < 2; ++tk) {
      v16h bk = kv[((size_t)h * (NNODES / 16) + (kb >> 4) + tk) * 32 + lane];
      v8f cz0 = {}, cz1 = {};
      v8f s0 = __builtin_amdgcn_wmma_f32_16x16x32_f16(false, qa0, false, bk,
                                                      (short)0, cz0, false, false);
      v8f s1 = __builtin_amdgcn_wmma_f32_16x16x32_f16(false, qa1, false, bk,
                                                      (short)0, cz1, false, false);
#pragma unroll
      for (int r = 0; r < 8; ++r) {
        s_tile[w][r + 8 * hf][tk * 16 + n]      = s0[r];
        s_tile[w][16 + r + 8 * hf][tk * 16 + n] = s1[r];
      }
    }
    __syncthreads();

    // ---- online softmax: each lane owns one of the 32 query rows ----
    {
      int row = lane;
      int gr = qbase + row;
      int qt = row >> 4, rloc = row & 15;
      float sv[32];
      float mnew = m_run;
#pragma unroll
      for (int j = 0; j < 32; ++j) {
        int gk = kb + j;
        unsigned char mb = mask[(size_t)gr * NNODES + gk];
        float add;
        if (mb) add = (mb == 1) ? ew0 : (mb == 2) ? ew1 : (mb == 3) ? ew2 : ew3;
        else    add = (gr == gk) ? ew3 : NEGV;
        float val = s_tile[w][row][j] + add;
        sv[j] = val;
        mnew = fmaxf(mnew, val);
      }
      float alpha = __expf(m_run - mnew);   // first block: exp(-inf-finite)=0
      float ssum = 0.f;
#pragma unroll
      for (int d = 0; d < 32; ++d) {
        float e = __expf(sv[d] - mnew);
        ssum += e;
        // scatter into A-fragment layout: elem d of row rloc
        int p = d >> 1, j = d & 1;
        int lane2 = rloc + (((p >> 3) & 1) << 4);
        int v2 = p & 7;
        ((_Float16*)&p_frag[w][qt][lane2])[2 * v2 + j] = (_Float16)e;
      }
      l_run = l_run * alpha + ssum;
      m_run = mnew;
      stat[w][row] = alpha;
    }
    __syncthreads();

    // ---- rescale O and accumulate O += P @ V ----
#pragma unroll
    for (int r = 0; r < 8; ++r) {
      float a0 = stat[w][r + 8 * hf];
      float a1 = stat[w][16 + r + 8 * hf];
      o00[r] *= a0; o01[r] *= a0;
      o10[r] *= a1; o11[r] *= a1;
    }
    v16h pa0 = p_frag[w][0][lane];
    v16h pa1 = p_frag[w][1][lane];
#pragma unroll
    for (int td = 0; td < 2; ++td) {
      v16h vb = vv[(((size_t)h * 2 + td) * (NNODES / 32) + (kb >> 5)) * 32 + lane];
      if (td == 0) {
        o00 = __builtin_amdgcn_wmma_f32_16x16x32_f16(false, pa0, false, vb,
                                                     (short)0, o00, false, false);
        o10 = __builtin_amdgcn_wmma_f32_16x16x32_f16(false, pa1, false, vb,
                                                     (short)0, o10, false, false);
      } else {
        o01 = __builtin_amdgcn_wmma_f32_16x16x32_f16(false, pa0, false, vb,
                                                     (short)0, o01, false, false);
        o11 = __builtin_amdgcn_wmma_f32_16x16x32_f16(false, pa1, false, vb,
                                                     (short)0, o11, false, false);
      }
    }
    __syncthreads();
  }

  // ---- finalize: O /= l ----
  stat[w][lane] = 1.f / l_run;
  __syncthreads();
#pragma unroll
  for (int r = 0; r < 8; ++r) {
    float i0 = stat[w][r + 8 * hf];
    float i1 = stat[w][16 + r + 8 * hf];
    int row0 = qbase + r + 8 * hf;
    int row1 = qbase + 16 + r + 8 * hf;
    out[(size_t)row0 * DMODEL + h * DHEAD + n]      = o00[r] * i0;
    out[(size_t)row0 * DMODEL + h * DHEAD + 16 + n] = o01[r] * i0;
    out[(size_t)row1 * DMODEL + h * DHEAD + n]      = o10[r] * i1;
    out[(size_t)row1 * DMODEL + h * DHEAD + 16 + n] = o11[r] * i1;
  }
}

// ---------------------------------------------------------------------------
// Host-side orchestration
// ---------------------------------------------------------------------------
extern "C" void kernel_launch(void* const* d_in, const int* in_sizes, int n_in,
                              void* d_out, int out_size, void* d_ws, size_t ws_size,
                              hipStream_t stream) {
  const float* agg    = (const float*)d_in[0];
  const int*   sid    = (const int*)d_in[1];
  const int*   eidx   = (const int*)d_in[2];
  const int*   etype  = (const int*)d_in[3];
  const float* spk    = (const float*)d_in[4];
  const float* W_in   = (const float*)d_in[5];
  const float* b_in   = (const float*)d_in[6];
  const float* edge_w = (const float*)d_in[7];
  const float* Wq     = (const float*)d_in[8];
  const float* bq     = (const float*)d_in[9];
  const float* Wk     = (const float*)d_in[10];
  const float* bk     = (const float*)d_in[11];
  const float* Wv     = (const float*)d_in[12];
  const float* bv     = (const float*)d_in[13];
  const float* in_w   = (const float*)d_in[14];
  const float* in_b   = (const float*)d_in[15];
  const float* out_w  = (const float*)d_in[16];
  const float* out_b  = (const float*)d_in[17];
  float* outp = (float*)d_out;

  const size_t MB = 1024 * 1024;
  const size_t KB = 1024;
  char* ws = (char*)d_ws;
  unsigned char* mask = (unsigned char*)(ws + 0);        // 16 MB
  float*    xc    = (float*)(ws + 16 * MB);              //  8 MB [N,512]
  float*    x     = (float*)(ws + 24 * MB);              //  4 MB [N,256]
  _Float16* qp    = (_Float16*)(ws + 28 * MB);           //  2 MB packed Q
  _Float16* kp    = (_Float16*)(ws + 30 * MB);           //  2 MB packed K
  _Float16* vp    = (_Float16*)(ws + 32 * MB);           //  2 MB packed V
  float*    o     = (float*)(ws + 34 * MB);              //  4 MB
  _Float16* wpin  = (_Float16*)(ws + 38 * MB);           //  256 KB packed W_in
  _Float16* wptmp = (_Float16*)(ws + 38 * MB + 256 * KB);//  128 KB packed scratch
  _Float16* wfpq  = (_Float16*)(ws + 38 * MB + 384 * KB);//  128 KB fused Wq
  _Float16* wfpk  = (_Float16*)(ws + 38 * MB + 512 * KB);
  _Float16* wfpv  = (_Float16*)(ws + 38 * MB + 640 * KB);
  float*    bfq   = (float*)(ws + 38 * MB + 768 * KB);   //  1 KB each
  float*    bfk   = (float*)(ws + 38 * MB + 769 * KB);
  float*    bfv   = (float*)(ws + 38 * MB + 770 * KB);
  float*    ew    = (float*)(ws + 38 * MB + 771 * KB);   //  32 B

  const float scale = 1.0f / sqrtf((float)DHEAD);

  auto gemm = [&](const float* A, const _Float16* Bp, const float* bias,
                  float* Cf, _Float16* Ch, int M, int K, int N, int mode, float scl) {
    int blocks = ((M / 16) * (N / 64)) / 4;
    gemm_wmma_kernel<<<blocks, 128, 0, stream>>>(A, Bp, bias, Cf, Ch, M, K, N, mode, scl);
  };
  auto packb = [&](const float* B, _Float16* Bp, int K, int N) {
    int blocks = ((K / 32) * (N / 16)) / 4;
    pack_b_kernel<<<blocks, 128, 0, stream>>>(B, Bp, K, N);
  };

  // prep
  softplus_kernel<<<1, 32, 0, stream>>>(edge_w, ew);
  clear_mask_kernel<<<16384, 256, 0, stream>>>((unsigned int*)mask);  // 16 MB
  scatter_edges_kernel<<<NEDGES / 256, 256, 0, stream>>>(eidx, etype, mask);
  build_xc_kernel<<<NNODES, 256, 0, stream>>>(agg, sid, spk, xc);

  // x = elu(xc @ W_in + b_in)
  packb(W_in, wpin, 512, DMODEL);
  gemm(xc, wpin, b_in, x, nullptr, NNODES, 512, DMODEL, 1, 1.f);

  for (int l = 0; l < 2; ++l) {
    const float* Wql = Wq + (size_t)l * 65536;
    const float* Wkl = Wk + (size_t)l * 65536;
    const float* Wvl = Wv + (size_t)l * 65536;
    const float* iw0 = in_w + ((size_t)l * 3 + 0) * 65536;
    const float* iw1 = in_w + ((size_t)l * 3 + 1) * 65536;
    const float* iw2 = in_w + ((size_t)l * 3 + 2) * 65536;

    // fused weights Wf = Wouter @ Win_proj, emitted directly as packed B frags
    packb(iw0, wptmp, DMODEL, DMODEL);
    gemm(Wql, wptmp, nullptr, nullptr, wfpq, DMODEL, DMODEL, DMODEL, 5, 1.f);
    packb(iw1, wptmp, DMODEL, DMODEL);
    gemm(Wkl, wptmp, nullptr, nullptr, wfpk, DMODEL, DMODEL, DMODEL, 5, 1.f);
    packb(iw2, wptmp, DMODEL, DMODEL);
    gemm(Wvl, wptmp, nullptr, nullptr, wfpv, DMODEL, DMODEL, DMODEL, 5, 1.f);
    // fused biases: bf = bouter @ Win_proj + bin_proj
    fused_bias_kernel<<<1, 256, 0, stream>>>(bq + (size_t)l * 256, iw0,
                                             in_b + ((size_t)l * 3 + 0) * 256, bfq);
    fused_bias_kernel<<<1, 256, 0, stream>>>(bk + (size_t)l * 256, iw1,
                                             in_b + ((size_t)l * 3 + 1) * 256, bfk);
    fused_bias_kernel<<<1, 256, 0, stream>>>(bv + (size_t)l * 256, iw2,
                                             in_b + ((size_t)l * 3 + 2) * 256, bfv);

    // projections straight into packed attention fragments
    gemm(x, wfpq, bfq, nullptr, qp, NNODES, DMODEL, DMODEL, 2, scale);
    gemm(x, wfpk, bfk, nullptr, kp, NNODES, DMODEL, DMODEL, 3, 1.f);
    gemm(x, wfpv, bfv, nullptr, vp, NNODES, DMODEL, DMODEL, 4, 1.f);

    // attention: 8 heads * 128 query blocks = 1024 waves, 4 waves / block
    attn_kernel<<<(NHEADS * (NNODES / 32)) / 4, 128, 0, stream>>>(
        qp, kp, vp, mask, ew + l * 4, o);

    // out projection (+ ELU between layers); last layer writes d_out
    packb(out_w + (size_t)l * 65536, wptmp, DMODEL, DMODEL);
    float* dst = (l == 1) ? outp : x;
    gemm(o, wptmp, out_b + (size_t)l * 256, dst, nullptr,
         NNODES, DMODEL, DMODEL, l == 0 ? 1 : 0, 1.f);
  }
}
